// FNSage_19567871001288
// MI455X (gfx1250) — matrix operand order repaired and checked by reading, not compiled, not run on hardware
//
#include <hip/hip_runtime.h>

typedef __bf16 bf16_t;
typedef __attribute__((ext_vector_type(16))) __bf16 v16bf;
typedef __attribute__((ext_vector_type(8)))  float  v8f;

// ---------------------------------------------------------------------------
// Utility: zero a buffer with float4 stores (grid-stride)
// ---------------------------------------------------------------------------
__global__ void zero4_kernel(float4* __restrict__ p, long n4) {
  long i = (long)blockIdx.x * blockDim.x + threadIdx.x;
  long stride = (long)gridDim.x * blockDim.x;
  float4 z = {0.0f, 0.0f, 0.0f, 0.0f};
  for (; i < n4; i += stride) p[i] = z;
}

// ---------------------------------------------------------------------------
// Degree: one thread per edge, atomic f32 add (counts stay in L2)
// ---------------------------------------------------------------------------
__global__ void degree_kernel(const int* __restrict__ dst, float* __restrict__ deg, int E) {
  int e = blockIdx.x * blockDim.x + threadIdx.x;
  if (e < E) atomicAdd(&deg[dst[e]], 1.0f);
}

// ---------------------------------------------------------------------------
// Edge aggregation: one wave32 per edge, 128 floats => float4 per lane.
// Gathered source rows are L2-resident (feature matrix = 25.6MB << 192MB L2).
// Prefetch a future edge's source row to hide gather latency.
// ---------------------------------------------------------------------------
__global__ void aggregate_kernel(const float* __restrict__ x, const int* __restrict__ src,
                                 const int* __restrict__ dst, float* __restrict__ agg, int E) {
  int gid = blockIdx.x * blockDim.x + threadIdx.x;
  int e = gid >> 5;
  int lane = threadIdx.x & 31;
  if (e >= E) return;
  int s = src[e];
  int d = dst[e];
  if (e + 8 < E) {
    int s2 = src[e + 8];
    __builtin_prefetch(x + (size_t)s2 * 128 + lane * 4, 0, 0);  // global_prefetch_b8
  }
  float4 v = *((const float4*)(x + (size_t)s * 128) + lane);
  float* a = agg + (size_t)d * 128 + lane * 4;
  atomicAdd(a + 0, v.x);
  atomicAdd(a + 1, v.y);
  atomicAdd(a + 2, v.z);
  atomicAdd(a + 3, v.w);
}

// ---------------------------------------------------------------------------
// Pre-split weights into bf16 hi/lo and permute into the WMMA B-fragment
// layout (16-bit B 32x16, this code's contiguous k order, same as A):
//   lane<16 : element e = W[k0 + e,       c0 + (lane&15)]
//   lane>=16: element e = W[k0 + 16 + e,  c0 + (lane&15)]
// Stored as [kt][ct][lane][16] contiguous bf16 => two b128 loads per lane.
// ---------------------------------------------------------------------------
__global__ void pack_weight_kernel(const float* __restrict__ w, unsigned short* __restrict__ hi_,
                                   unsigned short* __restrict__ lo_, int K, int Nout) {
  int t = blockIdx.x * blockDim.x + threadIdx.x;
  int CT = Nout >> 4;
  int total = (K >> 5) * CT * 32;
  if (t >= total) return;
  int lane = t & 31;
  int f = t >> 5;
  int ct = f % CT;
  int kt = f / CT;
  int kbase = kt * 32 + ((lane >> 4) << 4);
  int col = ct * 16 + (lane & 15);
  bf16_t* ph = (bf16_t*)hi_ + (size_t)t * 16;
  bf16_t* pl = (bf16_t*)lo_ + (size_t)t * 16;
#pragma unroll
  for (int e = 0; e < 16; ++e) {
    float v = w[(size_t)(kbase + e) * Nout + col];
    bf16_t h = (bf16_t)v;
    ph[e] = h;
    pl[e] = (bf16_t)(v - (float)h);
  }
}

// ---------------------------------------------------------------------------
// Load one 16x32 A fragment (row-major f32 source), scale rows, split bf16
// hi/lo. A layout (16-bit A 16x32, contiguous k order, same as B):
//   lane<16 : row=lane,    elements 0..15 = K k0..k0+15
//   lane>=16: row=lane-16, elements 0..15 = K k0+16..k0+31
// Caller passes p already offset (+16 floats for upper half-wave).
// ---------------------------------------------------------------------------
__device__ __forceinline__ void load_a_frag(const float* __restrict__ p, float scale,
                                            v16bf& hi, v16bf& lo) {
  const float4* q = (const float4*)p;
  float4 f0 = q[0], f1 = q[1], f2 = q[2], f3 = q[3];
  float v[16] = {f0.x, f0.y, f0.z, f0.w, f1.x, f1.y, f1.z, f1.w,
                 f2.x, f2.y, f2.z, f2.w, f3.x, f3.y, f3.z, f3.w};
#pragma unroll
  for (int e = 0; e < 16; ++e) {
    float s = v[e] * scale;
    bf16_t h = (bf16_t)s;
    hi[e] = h;
    lo[e] = (bf16_t)(s - (float)h);
  }
}

// ---------------------------------------------------------------------------
// Fused SAGE layer: out = relu( (agg/deg) @ Wl + x @ Wr + b )
// One wave32 computes a FULL 16 x NOUT row strip: A fragments are split to
// bf16 hi/lo ONCE per K-step and reused across all NOUT/16 column tiles.
// bf16x3 error compensation:  A@W ~= Ahi@Whi + Alo@Whi + Ahi@Wlo  (f32 acc).
// K = 128 fixed, NOUT templated => fully unrolled: 6 * 4 * (NOUT/16) WMMAs,
// NOUT/16 independent accumulator chains hide XDL latency.
// ---------------------------------------------------------------------------
template <int NOUT>
__global__ void __launch_bounds__(32) sage_wmma_kernel(
    const float* __restrict__ agg, const float* __restrict__ deg, const float* __restrict__ x,
    const unsigned short* __restrict__ wl_hi, const unsigned short* __restrict__ wl_lo,
    const unsigned short* __restrict__ wr_hi, const unsigned short* __restrict__ wr_lo,
    const float* __restrict__ bias, float* __restrict__ out) {
  constexpr int K = 128;
  constexpr int CT = NOUT >> 4;  // column tiles
  constexpr int KT = K >> 5;     // K steps of 32
  int lane = threadIdx.x;
  int r0 = blockIdx.x << 4;
  int mrow = r0 + (lane & 15);
  int koff = (lane >> 4) << 4;  // +16 floats for upper half-wave (A layout)
  float rscale = 1.0f / fmaxf(deg[mrow], 1.0f);
  const v16bf* BLH = (const v16bf*)wl_hi;
  const v16bf* BLL = (const v16bf*)wl_lo;
  const v16bf* BRH = (const v16bf*)wr_hi;
  const v16bf* BRL = (const v16bf*)wr_lo;

  v8f c[CT];
#pragma unroll
  for (int t = 0; t < CT; ++t) c[t] = {};

#pragma unroll
  for (int kt = 0; kt < KT; ++kt) {
    int k0 = kt << 5;
    const float* pa = agg + (size_t)mrow * K + k0 + koff;
    const float* px = x   + (size_t)mrow * K + k0 + koff;
    v16bf aah, aal, axh, axl;
    load_a_frag(pa, rscale, aah, aal);
    load_a_frag(px, 1.0f, axh, axl);
#pragma unroll
    for (int ct = 0; ct < CT; ++ct) {
      int f = (kt * CT + ct) * 32 + lane;
      v16bf blh = BLH[f];
      v16bf bll = BLL[f];
      v16bf brh = BRH[f];
      v16bf brl = BRL[f];
      c[ct] = __builtin_amdgcn_wmma_f32_16x16x32_bf16(false, aah, false, blh, (short)0, c[ct], false, false);
      c[ct] = __builtin_amdgcn_wmma_f32_16x16x32_bf16(false, aal, false, blh, (short)0, c[ct], false, false);
      c[ct] = __builtin_amdgcn_wmma_f32_16x16x32_bf16(false, aah, false, bll, (short)0, c[ct], false, false);
      c[ct] = __builtin_amdgcn_wmma_f32_16x16x32_bf16(false, axh, false, brh, (short)0, c[ct], false, false);
      c[ct] = __builtin_amdgcn_wmma_f32_16x16x32_bf16(false, axl, false, brh, (short)0, c[ct], false, false);
      c[ct] = __builtin_amdgcn_wmma_f32_16x16x32_bf16(false, axh, false, brl, (short)0, c[ct], false, false);
    }
  }

  // C layout: VGPR j => row (j or j+8); col = lane&15 (lanes 16-31 cover +8)
  int rbase = r0 + ((lane >> 4) << 3);
#pragma unroll
  for (int ct = 0; ct < CT; ++ct) {
    int col = (ct << 4) + (lane & 15);
    float bv = bias[col];
#pragma unroll
    for (int j = 0; j < 8; ++j) {
      float v = fmaxf(c[ct][j] + bv, 0.0f);  // relu (all three layers)
      out[(size_t)(rbase + j) * NOUT + col] = v;
    }
  }
}

// ---------------------------------------------------------------------------
// Final classifier: [N,64] @ [64,4] + b. N_CLASSES=4 doesn't fill a WMMA
// tile; trivial VALU cost (50K rows x 256 FMA).
// ---------------------------------------------------------------------------
__global__ void classifier_kernel(const float* __restrict__ h, const float* __restrict__ w,
                                  const float* __restrict__ b, float* __restrict__ out, int N) {
  int i = blockIdx.x * blockDim.x + threadIdx.x;
  if (i >= N) return;
  const float4* wv = (const float4*)w;
  float4 acc = *(const float4*)b;
  const float* hr = h + (size_t)i * 64;
#pragma unroll 8
  for (int k = 0; k < 64; ++k) {
    float hv = hr[k];
    float4 wk = wv[k];
    acc.x += hv * wk.x;
    acc.y += hv * wk.y;
    acc.z += hv * wk.z;
    acc.w += hv * wk.w;
  }
  ((float4*)out)[i] = acc;
}

// ---------------------------------------------------------------------------
extern "C" void kernel_launch(void* const* d_in, const int* in_sizes, int n_in,
                              void* d_out, int out_size, void* d_ws, size_t ws_size,
                              hipStream_t stream) {
  (void)n_in; (void)out_size; (void)ws_size;
  const float* x0    = (const float*)d_in[0];
  const int*   eix   = (const int*)d_in[1];
  // d_in[2] = edge_type: unused by the reference
  const float* w_l0  = (const float*)d_in[3];
  const float* w_r0  = (const float*)d_in[4];
  const float* b0    = (const float*)d_in[5];
  const float* w_l1  = (const float*)d_in[6];
  const float* w_r1  = (const float*)d_in[7];
  const float* b1    = (const float*)d_in[8];
  const float* w_l2  = (const float*)d_in[9];
  const float* w_r2  = (const float*)d_in[10];
  const float* b2    = (const float*)d_in[11];
  const float* w_out = (const float*)d_in[12];
  const float* b_out = (const float*)d_in[13];

  const int E = in_sizes[2];          // n_edges (edge_type length)
  const int N = in_sizes[0] / 128;    // n_nodes
  const int* src = eix;
  const int* dst = eix + E;

  // ---- workspace carve-up (256B aligned) ----
  char* base = (char*)d_ws;
  size_t off = 0;
  auto alloc = [&](size_t bytes) -> void* {
    void* p = base + off;
    off += (bytes + 255) & ~(size_t)255;
    return p;
  };
  float* deg = (float*)alloc((size_t)N * 4);
  float* agg = (float*)alloc((size_t)N * 128 * 4);
  float* hA  = (float*)alloc((size_t)N * 128 * 4);
  float* hB  = (float*)alloc((size_t)N * 128 * 4);
  const size_t F128 = (size_t)4 * 8 * 32 * 16 * 2;  // bytes: K=128,Nout=128 fragments
  const size_t F64  = (size_t)4 * 4 * 32 * 16 * 2;  // bytes: K=128,Nout=64 fragments
  unsigned short* wl0h = (unsigned short*)alloc(F128);
  unsigned short* wl0l = (unsigned short*)alloc(F128);
  unsigned short* wr0h = (unsigned short*)alloc(F128);
  unsigned short* wr0l = (unsigned short*)alloc(F128);
  unsigned short* wl1h = (unsigned short*)alloc(F128);
  unsigned short* wl1l = (unsigned short*)alloc(F128);
  unsigned short* wr1h = (unsigned short*)alloc(F128);
  unsigned short* wr1l = (unsigned short*)alloc(F128);
  unsigned short* wl2h = (unsigned short*)alloc(F64);
  unsigned short* wl2l = (unsigned short*)alloc(F64);
  unsigned short* wr2h = (unsigned short*)alloc(F64);
  unsigned short* wr2l = (unsigned short*)alloc(F64);

  // ---- degrees (shared by all layers) ----
  zero4_kernel<<<256, 256, 0, stream>>>((float4*)deg, (long)N / 4);
  degree_kernel<<<(E + 255) / 256, 256, 0, stream>>>(dst, deg, E);

  // ---- pack weights into bf16 hi/lo WMMA fragments ----
  pack_weight_kernel<<<4, 256, 0, stream>>>(w_l0, wl0h, wl0l, 128, 128);
  pack_weight_kernel<<<4, 256, 0, stream>>>(w_r0, wr0h, wr0l, 128, 128);
  pack_weight_kernel<<<4, 256, 0, stream>>>(w_l1, wl1h, wl1l, 128, 128);
  pack_weight_kernel<<<4, 256, 0, stream>>>(w_r1, wr1h, wr1l, 128, 128);
  pack_weight_kernel<<<2, 256, 0, stream>>>(w_l2, wl2h, wl2l, 128, 64);
  pack_weight_kernel<<<2, 256, 0, stream>>>(w_r2, wr2h, wr2l, 128, 64);

  const long agg4 = (long)N * 32;          // float4 count of agg buffer
  const int aggBlocks = (E + 7) / 8;       // 8 waves (edges) per 256-thread block
  const int strips = N / 16;               // 16-row strips

  // ---- layer 0 ----
  zero4_kernel<<<4096, 256, 0, stream>>>((float4*)agg, agg4);
  aggregate_kernel<<<aggBlocks, 256, 0, stream>>>(x0, src, dst, agg, E);
  sage_wmma_kernel<128><<<strips, 32, 0, stream>>>(agg, deg, x0, wl0h, wl0l, wr0h, wr0l, b0, hA);
  // ---- layer 1 ----
  zero4_kernel<<<4096, 256, 0, stream>>>((float4*)agg, agg4);
  aggregate_kernel<<<aggBlocks, 256, 0, stream>>>(hA, src, dst, agg, E);
  sage_wmma_kernel<128><<<strips, 32, 0, stream>>>(agg, deg, hA, wl1h, wl1l, wr1h, wr1l, b1, hB);
  // ---- layer 2 (output dim 64; reuse hA as h3) ----
  zero4_kernel<<<4096, 256, 0, stream>>>((float4*)agg, agg4);
  aggregate_kernel<<<aggBlocks, 256, 0, stream>>>(hB, src, dst, agg, E);
  sage_wmma_kernel<64><<<strips, 32, 0, stream>>>(agg, deg, hB, wl2h, wl2l, wr2h, wr2l, b2, hA);
  // ---- classifier ----
  classifier_kernel<<<(N + 255) / 256, 256, 0, stream>>>(hA, w_out, b_out, (float*)d_out, N);
}
